// SimpleRNN_73443940761963
// MI455X (gfx1250) — matrix operand order repaired and checked by reading, not compiled, not run on hardware
//
#include <hip/hip_runtime.h>
#include <math.h>

#define VOCAB   32000
#define HIDDEN  2048
#define NLAYERS 6
#define BATCH   512

// Tiling
#define BM   64     // output rows per workgroup
#define BN   128    // output cols per workgroup
#define KC   64     // K chunk = two bf16 WMMA K-steps per stage
#define LDSK 72     // padded K stride in LDS (bf16 elems) = 144B: 16B-aligned, bank-conflict-free

typedef __attribute__((ext_vector_type(16))) __bf16 v16bf;
typedef __attribute__((ext_vector_type(4)))  __bf16 v4bf;
typedef __attribute__((ext_vector_type(8)))  float  v8f;
typedef __attribute__((ext_vector_type(4)))  float  f32x4;
typedef __attribute__((ext_vector_type(4)))  unsigned int u32x4;

union FragCast { u32x4 u[2]; v16bf v; };

// A fragment, 16x32 bf16 (ISA 7.12.2): lanes 0-15 -> row M=lane, K = {0..7, 16..23};
// lanes 16-31 -> row M=lane-16, K = {8..15, 24..31}. v16bf element e maps to
// k = kb + (e<8 ? e : e+8), kb = koff + (lane<16 ? 0 : 8).
__device__ __forceinline__ v16bf ld_frag_a(const __bf16* lA, int tm, int koff, int lane) {
  const int row = tm + (lane & 15);
  const int kb  = koff + ((lane < 16) ? 0 : 8);
  FragCast f;
  f.u[0] = *(const u32x4*)(lA + row * LDSK + kb);        // K = kb .. kb+7
  f.u[1] = *(const u32x4*)(lA + row * LDSK + kb + 16);   // K = kb+16 .. kb+23
  return f.v;
}

// B fragment, 32x16 bf16: lanes 0-15 -> col N=lane, K = {0..7, 16..23}; lanes 16-31 -> col N=lane-16,
// K = {8..15, 24..31}. lB is stored transposed [n][k] so this is two 16B reads per lane.
__device__ __forceinline__ v16bf ld_frag_b(const __bf16* lB, int tn, int koff, int lane) {
  const int col = tn + (lane & 15);
  const int kb  = koff + ((lane < 16) ? 0 : 8);
  FragCast f;
  f.u[0] = *(const u32x4*)(lB + col * LDSK + kb);        // K = kb .. kb+7
  f.u[1] = *(const u32x4*)(lB + col * LDSK + kb + 16);   // K = kb+16 .. kb+23
  return f.v;
}

// Global -> registers for one K-chunk (software pipeline stage).
// A: 64x64 f32 = 1024 float4, 4/thread.  B: 64x128 f32 = 2048 float4, 8/thread.
__device__ __forceinline__ void load_chunk(const float* __restrict__ state_in,
                                           const float* __restrict__ hh,
                                           int bm, int bn, int k0, int tid,
                                           f32x4 ra[4], f32x4 rb[8]) {
#pragma unroll
  for (int r = 0; r < 4; ++r) {
    const int idx = tid + (r << 8);
    const int row = idx >> 4;                // 0..63
    const int c4  = (idx & 15) << 2;         // 0,4,..,60
    ra[r] = *(const f32x4*)(state_in + (size_t)(bm + row) * HIDDEN + k0 + c4);
  }
#pragma unroll
  for (int r = 0; r < 8; ++r) {
    const int idx = tid + (r << 8);
    const int k   = idx >> 5;                // 0..63
    const int n4  = (idx & 31) << 2;         // 0,4,..,124
    const float* gp = hh + (size_t)(k0 + k) * HIDDEN + bn + n4;
    rb[r] = *(const f32x4*)gp;
    if (k0 + KC < HIDDEN)                    // global_prefetch_b8: pull chunk after next into L2
      __builtin_prefetch(gp + (size_t)KC * HIDDEN, 0, 1);
  }
}

// Registers -> LDS with f32->bf16 conversion (B stored transposed [n][k]).
__device__ __forceinline__ void store_chunk(__bf16* lA, __bf16* lB, int tid,
                                            const f32x4 ra[4], const f32x4 rb[8]) {
#pragma unroll
  for (int r = 0; r < 4; ++r) {
    const int idx = tid + (r << 8);
    const int row = idx >> 4;
    const int c4  = (idx & 15) << 2;
    v4bf b;
    b.x = (__bf16)ra[r].x; b.y = (__bf16)ra[r].y;
    b.z = (__bf16)ra[r].z; b.w = (__bf16)ra[r].w;
    *(v4bf*)(lA + row * LDSK + c4) = b;
  }
#pragma unroll
  for (int r = 0; r < 8; ++r) {
    const int idx = tid + (r << 8);
    const int k   = idx >> 5;
    const int n4  = (idx & 31) << 2;
    lB[(n4 + 0) * LDSK + k] = (__bf16)rb[r].x;
    lB[(n4 + 1) * LDSK + k] = (__bf16)rb[r].y;
    lB[(n4 + 2) * LDSK + k] = (__bf16)rb[r].z;
    lB[(n4 + 3) * LDSK + k] = (__bf16)rb[r].w;
  }
}

// One RNN layer: state_out = tanh(state_in + (state_in @ hh) * ih[token])
__global__ __launch_bounds__(256) void rnn_layer_kernel(
    const float* __restrict__ state_in,   // [BATCH][HIDDEN]
    const float* __restrict__ hh,         // [HIDDEN][HIDDEN]   (this layer)
    const float* __restrict__ ih,         // [VOCAB][HIDDEN]    (this layer)
    const int*   __restrict__ token,      // [BATCH]
    float*       __restrict__ state_out)  // [BATCH][HIDDEN]
{
  __shared__ __align__(16) __bf16 lA[BM * LDSK];   // [m][k], padded
  __shared__ __align__(16) __bf16 lB[BN * LDSK];   // [n][k], transposed + padded

  const int tid  = threadIdx.x;
  const int lane = tid & 31;
  const int wave = tid >> 5;     // 8 waves
  const int wm   = wave & 1;     // 2 wave-rows  -> 32 output rows each
  const int wn   = wave >> 1;    // 4 wave-cols  -> 32 output cols each
  const int bm   = blockIdx.x * BM;
  const int bn   = blockIdx.y * BN;

  v8f acc[2][2] = {};            // 2x2 tiles of 16x16 f32 per wave

  f32x4 ra[4], rb[8];
  load_chunk(state_in, hh, bm, bn, 0, tid, ra, rb);   // prologue: chunk 0 -> regs

  for (int k0 = 0; k0 < HIDDEN; k0 += KC) {
    // regs (chunk k0) -> LDS
    store_chunk(lA, lB, tid, ra, rb);
    __syncthreads();

    // prefetch chunk k0+KC into registers; its latency hides under the 8 WMMAs below
    if (k0 + KC < HIDDEN)
      load_chunk(state_in, hh, bm, bn, k0 + KC, tid, ra, rb);

    // compute: 2 K-steps x (2x2 tiles) = 8 WMMAs per stage
#pragma unroll
    for (int t = 0; t < KC / 32; ++t) {
      const int ko = t * 32;
      v16bf a0 = ld_frag_a(lA, wm * 32 + 0,  ko, lane);
      v16bf a1 = ld_frag_a(lA, wm * 32 + 16, ko, lane);
      v16bf b0 = ld_frag_b(lB, wn * 32 + 0,  ko, lane);
      v16bf b1 = ld_frag_b(lB, wn * 32 + 16, ko, lane);

      acc[0][0] = __builtin_amdgcn_wmma_f32_16x16x32_bf16(false, a0, false, b0,
                                                          (short)0, acc[0][0], false, false);
      acc[0][1] = __builtin_amdgcn_wmma_f32_16x16x32_bf16(false, a0, false, b1,
                                                          (short)0, acc[0][1], false, false);
      acc[1][0] = __builtin_amdgcn_wmma_f32_16x16x32_bf16(false, a1, false, b0,
                                                          (short)0, acc[1][0], false, false);
      acc[1][1] = __builtin_amdgcn_wmma_f32_16x16x32_bf16(false, a1, false, b1,
                                                          (short)0, acc[1][1], false, false);
    }
    __syncthreads();
  }

  // ---- epilogue: C layout (VGPR r: M=r (+8 for lanes>=16), N=lane&15) ----
  // out[b][h] = tanh(state_in[b][h] + acc * ih[token[b]][h])
#pragma unroll
  for (int i = 0; i < 2; ++i) {
#pragma unroll
    for (int j = 0; j < 2; ++j) {
      const int n = bn + wn * 32 + j * 16 + (lane & 15);
#pragma unroll
      for (int r = 0; r < 8; ++r) {
        const int m   = bm + wm * 32 + i * 16 + r + ((lane >> 4) << 3);
        const int tok = token[m];
        const float g = ih[(size_t)tok * HIDDEN + n];
        const float s = state_in[(size_t)m * HIDDEN + n];
        state_out[(size_t)m * HIDDEN + n] = tanhf(s + acc[i][j][r] * g);
      }
    }
  }
}

extern "C" void kernel_launch(void* const* d_in, const int* in_sizes, int n_in,
                              void* d_out, int out_size, void* d_ws, size_t ws_size,
                              hipStream_t stream) {
  // setup_inputs() order: state f32[512,2048], token i32[512],
  //                       ih f32[6,32000,2048], hh f32[6,2048,2048]
  const float* state = (const float*)d_in[0];
  const int*   token = (const int*)d_in[1];
  const float* ih    = (const float*)d_in[2];
  const float* hh    = (const float*)d_in[3];
  float* out = (float*)d_out;

  // ping-pong state buffers in workspace (2 * 512*2048 f32 = 8 MB)
  float* ws0 = (float*)d_ws;
  float* ws1 = ws0 + (size_t)BATCH * HIDDEN;

  dim3 grid(BATCH / BM, HIDDEN / BN);   // 8 x 16 = 128 workgroups / layer
  dim3 block(256);

  const float* cur = state;
  for (int l = 0; l < NLAYERS; ++l) {
    float* dst = (l == NLAYERS - 1) ? out : ((l & 1) ? ws1 : ws0);
    hipLaunchKernelGGL(rnn_layer_kernel, grid, block, 0, stream,
                       cur,
                       hh + (size_t)l * HIDDEN * HIDDEN,
                       ih + (size_t)l * VOCAB * HIDDEN,
                       token, dst);
    cur = dst;
  }

  (void)in_sizes; (void)n_in; (void)out_size; (void)ws_size;
}
